// VQVAE_71330816852854
// MI455X (gfx1250) — matrix operand order repaired and checked by reading, not compiled, not run on hardware
//
#include <hip/hip_runtime.h>

typedef __attribute__((ext_vector_type(16))) __bf16         v16bf;
typedef __attribute__((ext_vector_type(16))) unsigned short v16u;
typedef __attribute__((ext_vector_type(8)))  unsigned short u16x8;
typedef __attribute__((ext_vector_type(8)))  float          v8f;

enum { MODE_BIAS = 0, MODE_RELU = 1, MODE_PRELU = 2, MODE_RES = 3 };

constexpr int clog2(int v) { int s = 0; while ((1 << s) < v) ++s; return s; }

__device__ __forceinline__ unsigned short f2bf(float f) {
    unsigned int u = __float_as_uint(f);
    u += 0x7fffu + ((u >> 16) & 1u);          // round-to-nearest-even
    return (unsigned short)(u >> 16);
}

// ---------------- f32 -> bf16 elementwise (activations) ----------------
__global__ void cvt_bf16_kernel(const float* __restrict__ s,
                                unsigned short* __restrict__ d, int n) {
    int i = blockIdx.x * blockDim.x + threadIdx.x;
    if (i < n) d[i] = f2bf(s[i]);
}

// ---------------- f32 -> bf16 weight pack into WMMA B-fragment order -------
// src: K x cout row-major.  dst[((kc*cout + n) << 5) + (h << 4) + e]
//   = src[(kc*32 + (e < 8 ? 8h + e : 16 + 8h + e - 8))*cout + n]
// => each lane's 16 B-fragment elements are 32 contiguous bytes.
__global__ void pack_w_kernel(const float* __restrict__ s,
                              unsigned short* __restrict__ d,
                              int total, int coutShift) {
    int i = blockIdx.x * blockDim.x + threadIdx.x;
    if (i >= total) return;
    int k = i >> coutShift;
    int n = i & ((1 << coutShift) - 1);
    int kc = k >> 5, w = k & 31;
    int h = (w >> 3) & 1, e = (w & 7) | ((w >> 4) << 3);
    d[(((kc << coutShift) + n) << 5) + (h << 4) + e] = f2bf(s[i]);
}

// A fragment: two contiguous 16-byte loads (ap already includes +8*h)
__device__ __forceinline__ v16u load_afrag(const unsigned short* ap) {
    u16x8 lo = *(const u16x8*)(ap);
    u16x8 hi = *(const u16x8*)(ap + 16);
    return __builtin_shufflevector(lo, hi, 0,1,2,3,4,5,6,7,8,9,10,11,12,13,14,15);
}

__device__ __forceinline__ v16u zero_frag() {
    v16u z;
#pragma unroll
    for (int e = 0; e < 16; ++e) z[e] = 0;
    return z;
}

template<int MODE>
__device__ __forceinline__ float apply_mode(float v, float al, float g, float be,
                                            float mn, float iv, float res) {
    if (MODE == MODE_RELU)       return fmaxf(v, 0.f);
    if (MODE == MODE_PRELU)      return (v >= 0.f) ? v : al * v;
    if (MODE == MODE_RES) {
        v = (v - mn) * (g * iv) + be;
        v = (v >= 0.f) ? v : al * v;
        return fmaxf(res + v, 0.f);
    }
    return v;
}

// ---------------- conv3d implicit GEMM, 32x32 tile per wave ----------------
// All shape parameters are compile-time: shifts fold to immediates, the
// channel-chunk loop fully unrolls into a dense v_wmma stream.
template<int SI, int SO, int CIN, int COUT, int KS, int STRIDE, int MODE>
__global__ __launch_bounds__(32) void conv_wmma_t(
    const unsigned short* __restrict__ act, const unsigned short* __restrict__ wpk,
    const float* __restrict__ bias,  const float* __restrict__ gamma,
    const float* __restrict__ beta,  const float* __restrict__ mean,
    const float* __restrict__ var,   const float* __restrict__ alpha,
    const float* __restrict__ resid, float* __restrict__ out)
{
    constexpr int siSh  = clog2(SI);
    constexpr int soSh  = clog2(SO);
    constexpr int cinSh = clog2(CIN);
    constexpr int coutSh = clog2(COUT);
    constexpr int soMask = SO - 1;

    const int lane = threadIdx.x;
    const int lm = lane & 15;
    const int h  = lane >> 4;
    const int m32 = blockIdx.x * 32;
    const int n32 = blockIdx.y * 32;

    const int m0 = m32 + lm,      m1 = m32 + 16 + lm;
    const int x0 = m0 & soMask,   y0 = (m0 >> soSh) & soMask,   z0 = m0 >> (2*soSh);
    const int x1 = m1 & soMask,   y1 = (m1 >> soSh) & soMask,   z1 = m1 >> (2*soSh);

    const int bofs = (((n32 + lm) << 5) + (h << 4));   // B lane offset (elems)
    const int h8 = 8 * h;

    v8f acc00 = {0.f,0.f,0.f,0.f,0.f,0.f,0.f,0.f};
    v8f acc01 = acc00, acc10 = acc00, acc11 = acc00;

    int kc = 0;
#pragma unroll 1
    for (int kd = 0; kd < KS; ++kd) {
        const int iz0 = z0 * STRIDE + kd - 1, iz1 = z1 * STRIDE + kd - 1;
        const bool zr0 = (unsigned)iz0 < (unsigned)SI;
        const bool zr1 = (unsigned)iz1 < (unsigned)SI;
#pragma unroll 1
        for (int kh = 0; kh < KS; ++kh) {
            const int iy0 = y0 * STRIDE + kh - 1, iy1 = y1 * STRIDE + kh - 1;
            const bool yr0 = zr0 && ((unsigned)iy0 < (unsigned)SI);
            const bool yr1 = zr1 && ((unsigned)iy1 < (unsigned)SI);
#pragma unroll 1
            for (int kw = 0; kw < KS; ++kw) {
                const int ix0 = x0 * STRIDE + kw - 1, ix1 = x1 * STRIDE + kw - 1;
                const bool inr0 = yr0 && ((unsigned)ix0 < (unsigned)SI);
                const bool inr1 = yr1 && ((unsigned)ix1 < (unsigned)SI);
                const unsigned short* ap0 = act +
                    (((((iz0 << siSh) + iy0) << siSh) + ix0) << cinSh) + h8;
                const unsigned short* ap1 = act +
                    (((((iz1 << siSh) + iy1) << siSh) + ix1) << cinSh) + h8;
#pragma unroll
                for (int cb = 0; cb < CIN; cb += 32, ++kc) {
                    v16u au0, au1;
                    if (inr0) au0 = load_afrag(ap0 + cb); else au0 = zero_frag();
                    if (inr1) au1 = load_afrag(ap1 + cb); else au1 = zero_frag();
                    const unsigned short* wb = wpk + (kc << (coutSh + 5)) + bofs;
                    __builtin_prefetch(wb + (COUT << 5), 0, 0);   // next k-chunk
                    v16u bu0 = *(const v16u*)(wb);
                    v16u bu1 = *(const v16u*)(wb + 512);
                    v16bf a0 = __builtin_bit_cast(v16bf, au0);
                    v16bf a1 = __builtin_bit_cast(v16bf, au1);
                    v16bf b0 = __builtin_bit_cast(v16bf, bu0);
                    v16bf b1 = __builtin_bit_cast(v16bf, bu1);
                    acc00 = __builtin_amdgcn_wmma_f32_16x16x32_bf16(false, a0, false, b0, (short)0, acc00, false, false);
                    acc01 = __builtin_amdgcn_wmma_f32_16x16x32_bf16(false, a0, false, b1, (short)0, acc01, false, false);
                    acc10 = __builtin_amdgcn_wmma_f32_16x16x32_bf16(false, a1, false, b0, (short)0, acc10, false, false);
                    acc11 = __builtin_amdgcn_wmma_f32_16x16x32_bf16(false, a1, false, b1, (short)0, acc11, false, false);
                }
            }
        }
    }

    // ---------------- epilogue ----------------
    const int n0 = n32 + lm, n1 = n0 + 16;
    float bs0 = bias[n0], bs1 = bias[n1];
    float g0=0,be0=0,mn0=0,iv0=0,al0=0, g1=0,be1=0,mn1=0,iv1=0,al1=0;
    if (MODE == MODE_RES) {
        g0 = gamma[n0]; be0 = beta[n0]; mn0 = mean[n0];
        iv0 = rsqrtf(var[n0] + 1e-3f); al0 = alpha[n0];
        g1 = gamma[n1]; be1 = beta[n1]; mn1 = mean[n1];
        iv1 = rsqrtf(var[n1] + 1e-3f); al1 = alpha[n1];
    } else if (MODE == MODE_PRELU) {
        al0 = alpha[n0]; al1 = alpha[n1];
    }
#pragma unroll
    for (int r = 0; r < 8; ++r) {
        int row0 = m32 + 8*h + r;
        int row1 = row0 + 16;
        float r00=0.f, r01=0.f, r10=0.f, r11=0.f;
        if (MODE == MODE_RES) {
            r00 = resid[(row0 << coutSh) + n0]; r01 = resid[(row0 << coutSh) + n1];
            r10 = resid[(row1 << coutSh) + n0]; r11 = resid[(row1 << coutSh) + n1];
        }
        out[(row0 << coutSh) + n0] = apply_mode<MODE>(acc00[r] + bs0, al0, g0, be0, mn0, iv0, r00);
        out[(row0 << coutSh) + n1] = apply_mode<MODE>(acc01[r] + bs1, al1, g1, be1, mn1, iv1, r01);
        out[(row1 << coutSh) + n0] = apply_mode<MODE>(acc10[r] + bs0, al0, g0, be0, mn0, iv0, r10);
        out[(row1 << coutSh) + n1] = apply_mode<MODE>(acc11[r] + bs1, al1, g1, be1, mn1, iv1, r11);
    }
}

// ---------------- transpose conv 4x4x4 stride-2, parity-split, 32x32 tile ---
template<int SH, int CIN, int COUT>
__global__ __launch_bounds__(32) void convt_wmma_t(
    const unsigned short* __restrict__ act, const unsigned short* __restrict__ wpk,
    const float* __restrict__ bias, float* __restrict__ out)
{
    constexpr int shSh   = clog2(SH);
    constexpr int cinSh  = clog2(CIN);
    constexpr int coutSh = clog2(COUT);
    constexpr int shMask = SH - 1;
    constexpr int chunks = CIN >> 5;

    const int lane = threadIdx.x;
    const int lm = lane & 15;
    const int h  = lane >> 4;
    const int pc = blockIdx.z;
    const int pz = (pc >> 2) & 1, py = (pc >> 1) & 1, px = pc & 1;
    const int m32 = blockIdx.x * 32;
    const int n32 = blockIdx.y * 32;

    const int ms0 = m32 + lm, ms1 = m32 + 16 + lm;
    const int sx0 = ms0 & shMask, sy0 = (ms0 >> shSh) & shMask, sz0 = ms0 >> (2*shSh);
    const int sx1 = ms1 & shMask, sy1 = (ms1 >> shSh) & shMask, sz1 = ms1 >> (2*shSh);

    const int bofs = (((n32 + lm) << 5) + (h << 4));
    const int h8 = 8 * h;

    v8f acc00 = {0.f,0.f,0.f,0.f,0.f,0.f,0.f,0.f};
    v8f acc01 = acc00, acc10 = acc00, acc11 = acc00;

#pragma unroll 1
    for (int bd = 0; bd < 2; ++bd) {
        const int iz0 = sz0 + pz + bd - 1, iz1 = sz1 + pz + bd - 1;
        const bool zr0 = (unsigned)iz0 < (unsigned)SH;
        const bool zr1 = (unsigned)iz1 < (unsigned)SH;
        const int ktd = pz + 2 * bd;
#pragma unroll 1
        for (int bh = 0; bh < 2; ++bh) {
            const int iy0 = sy0 + py + bh - 1, iy1 = sy1 + py + bh - 1;
            const bool yr0 = zr0 && ((unsigned)iy0 < (unsigned)SH);
            const bool yr1 = zr1 && ((unsigned)iy1 < (unsigned)SH);
            const int kth = py + 2 * bh;
#pragma unroll 1
            for (int bw = 0; bw < 2; ++bw) {
                const int ix0 = sx0 + px + bw - 1, ix1 = sx1 + px + bw - 1;
                const bool inr0 = yr0 && ((unsigned)ix0 < (unsigned)SH);
                const bool inr1 = yr1 && ((unsigned)ix1 < (unsigned)SH);
                const int ktw = px + 2 * bw;
                const unsigned short* ap0 = act +
                    (((((iz0 << shSh) + iy0) << shSh) + ix0) << cinSh) + h8;
                const unsigned short* ap1 = act +
                    (((((iz1 << shSh) + iy1) << shSh) + ix1) << cinSh) + h8;
                int kc = ((ktd * 4 + kth) * 4 + ktw) * chunks;
#pragma unroll
                for (int cb = 0; cb < CIN; cb += 32, ++kc) {
                    v16u au0, au1;
                    if (inr0) au0 = load_afrag(ap0 + cb); else au0 = zero_frag();
                    if (inr1) au1 = load_afrag(ap1 + cb); else au1 = zero_frag();
                    const unsigned short* wb = wpk + (kc << (coutSh + 5)) + bofs;
                    __builtin_prefetch(wb + (COUT << 5), 0, 0);
                    v16u bu0 = *(const v16u*)(wb);
                    v16u bu1 = *(const v16u*)(wb + 512);
                    v16bf a0 = __builtin_bit_cast(v16bf, au0);
                    v16bf a1 = __builtin_bit_cast(v16bf, au1);
                    v16bf b0 = __builtin_bit_cast(v16bf, bu0);
                    v16bf b1 = __builtin_bit_cast(v16bf, bu1);
                    acc00 = __builtin_amdgcn_wmma_f32_16x16x32_bf16(false, a0, false, b0, (short)0, acc00, false, false);
                    acc01 = __builtin_amdgcn_wmma_f32_16x16x32_bf16(false, a0, false, b1, (short)0, acc01, false, false);
                    acc10 = __builtin_amdgcn_wmma_f32_16x16x32_bf16(false, a1, false, b0, (short)0, acc10, false, false);
                    acc11 = __builtin_amdgcn_wmma_f32_16x16x32_bf16(false, a1, false, b1, (short)0, acc11, false, false);
                }
            }
        }
    }

    const int n0 = n32 + lm, n1 = n0 + 16;
    const float bs0 = bias[n0], bs1 = bias[n1];
#pragma unroll
    for (int r = 0; r < 8; ++r) {
        int msr0 = m32 + 8*h + r;
        int msr1 = msr0 + 16;
        int sx2 = msr0 & shMask, sy2 = (msr0 >> shSh) & shMask, sz2 = msr0 >> (2*shSh);
        int sx3 = msr1 & shMask, sy3 = (msr1 >> shSh) & shMask, sz3 = msr1 >> (2*shSh);
        int mf0 = ((((2*sz2+pz) << (shSh+1)) + (2*sy2+py)) << (shSh+1)) + (2*sx2+px);
        int mf1 = ((((2*sz3+pz) << (shSh+1)) + (2*sy3+py)) << (shSh+1)) + (2*sx3+px);
        float v00 = fmaxf(acc00[r] + bs0, 0.f), v01 = fmaxf(acc01[r] + bs1, 0.f);
        float v10 = fmaxf(acc10[r] + bs0, 0.f), v11 = fmaxf(acc11[r] + bs1, 0.f);
        out[(mf0 << coutSh) + n0] = v00;
        out[(mf0 << coutSh) + n1] = v01;
        out[(mf1 << coutSh) + n0] = v10;
        out[(mf1 << coutSh) + n1] = v11;
    }
}

// ---------------- first downsample conv: cin = 1 (tiny, VALU) ---------------
__global__ void conv_d0_kernel(const float* __restrict__ in,
                               const float* __restrict__ w,
                               const float* __restrict__ b,
                               float* __restrict__ out)
{
    int t = blockIdx.x * blockDim.x + threadIdx.x;   // 32768 voxels * 32 ch
    if (t >= 32768 * 32) return;
    int n = t & 31;
    int m = t >> 5;
    int x = m & 31, y = (m >> 5) & 31, z = m >> 10;
    float acc = 0.f;
    for (int kd = 0; kd < 4; ++kd) {
        int iz = 2 * z + kd - 1; if (iz < 0 || iz >= 64) continue;
        for (int kh = 0; kh < 4; ++kh) {
            int iy = 2 * y + kh - 1; if (iy < 0 || iy >= 64) continue;
            for (int kw = 0; kw < 4; ++kw) {
                int ix = 2 * x + kw - 1; if (ix < 0 || ix >= 64) continue;
                acc += in[(iz * 64 + iy) * 64 + ix] *
                       w[((kd * 4 + kh) * 4 + kw) * 32 + n];
            }
        }
    }
    out[t] = fmaxf(acc + b[n], 0.f);
}

// ---------------- final transpose conv: cout = 1 (tiny, VALU) ---------------
__global__ void convt_c1_kernel(const float* __restrict__ in,
                                const float* __restrict__ w,
                                const float* __restrict__ b,
                                float* __restrict__ out)
{
    int m = blockIdx.x * blockDim.x + threadIdx.x;   // 64^3 outputs
    if (m >= 262144) return;
    int x = m & 63, y = (m >> 6) & 63, z = m >> 12;
    int sx = x >> 1, px = x & 1;
    int sy = y >> 1, py = y & 1;
    int sz = z >> 1, pz = z & 1;
    float acc = b[0];
    for (int tap = 0; tap < 8; ++tap) {
        int bd = (tap >> 2) & 1, bh = (tap >> 1) & 1, bw = tap & 1;
        int iz = sz + pz + bd - 1;
        int iy = sy + py + bh - 1;
        int ix = sx + px + bw - 1;
        if (iz < 0 || iz >= 32 || iy < 0 || iy >= 32 || ix < 0 || ix >= 32) continue;
        int ktd = pz + 2 * bd, kth = py + 2 * bh, ktw = px + 2 * bw;
        const float* ip = in + ((iz * 32 + iy) * 32 + ix) * 32;
        const float* wp = w + ((ktd * 4 + kth) * 4 + ktw) * 32;
        for (int c = 0; c < 32; ++c) acc += ip[c] * wp[c];
    }
    out[m] = acc;
}

// ---------------- VQ: forward of z + sg(q - z) is exactly q -----------------
__global__ void vq_kernel(const float* __restrict__ z,
                          const float* __restrict__ emb,   // [64][512]
                          float* __restrict__ q)
{
    int m = blockIdx.x * blockDim.x + threadIdx.x;   // 512 positions
    if (m >= 512) return;
    float zr[64];
    for (int c = 0; c < 64; ++c) zr[c] = z[m * 64 + c];
    int best = 0; float bd = 3.4e38f;
    for (int j = 0; j < 512; ++j) {
        float d = 0.f;
        for (int c = 0; c < 64; ++c) {
            float t = zr[c] - emb[c * 512 + j];
            d += t * t;
        }
        if (d < bd) { bd = d; best = j; }
    }
    for (int c = 0; c < 64; ++c) q[m * 64 + c] = emb[c * 512 + best];
}

// ============================ host orchestration ============================
static void cvt_launch(const float* src, unsigned short* dst, int n, hipStream_t st) {
    cvt_bf16_kernel<<<(n + 255) / 256, 256, 0, st>>>(src, dst, n);
}
static void pack_launch(const float* src, unsigned short* dst, int total,
                        int coutSh, hipStream_t st) {
    pack_w_kernel<<<(total + 255) / 256, 256, 0, st>>>(src, dst, total, coutSh);
}

template<int SI, int SO, int CIN, int COUT, int KS, int STRIDE, int MODE>
static void conv_launch(const float* in, const float* w, const float* bias,
                        const float* g, const float* be, const float* mn,
                        const float* vr, const float* al, const float* resid,
                        float* outp, unsigned short* Hb, unsigned short* Wb,
                        hipStream_t st) {
    cvt_launch(in, Hb, SI * SI * SI * CIN, st);
    pack_launch(w, Wb, KS * KS * KS * CIN * COUT, clog2(COUT), st);
    dim3 grid((SO * SO * SO) / 32, COUT / 32);
    conv_wmma_t<SI, SO, CIN, COUT, KS, STRIDE, MODE>
        <<<grid, 32, 0, st>>>(Hb, Wb, bias, g, be, mn, vr, al, resid, outp);
}

template<int SH, int CIN, int COUT>
static void convt_launch(const float* in, const float* w, const float* bias,
                         float* outp, unsigned short* Hb, unsigned short* Wb,
                         hipStream_t st) {
    cvt_launch(in, Hb, SH * SH * SH * CIN, st);
    pack_launch(w, Wb, 64 * CIN * COUT, clog2(COUT), st);
    dim3 grid((SH * SH * SH) / 32, COUT / 32, 8);
    convt_wmma_t<SH, CIN, COUT><<<grid, 32, 0, st>>>(Hb, Wb, bias, outp);
}

template<int S, int C>
static void resblock(void* const* d_in, unsigned short* Hb, unsigned short* Wb,
                     float*& cur, float*& t1, float*& t2, int base, hipStream_t st) {
    auto P = [&](int i) { return (const float*)d_in[i]; };
    conv_launch<S, S, C, C, 3, 1, MODE_RELU>(
        cur, P(base), P(base + 1),
        nullptr, nullptr, nullptr, nullptr, nullptr, nullptr, t1, Hb, Wb, st);
    conv_launch<S, S, C, C, 3, 1, MODE_RES>(
        t1, P(base + 2), P(base + 3),
        P(base + 4), P(base + 5), P(base + 6), P(base + 7), P(base + 8),
        cur, t2, Hb, Wb, st);
    float* tmp = cur; cur = t2; t2 = tmp;
}

extern "C" void kernel_launch(void* const* d_in, const int* in_sizes, int n_in,
                              void* d_out, int out_size, void* d_ws, size_t ws_size,
                              hipStream_t stream)
{
    (void)in_sizes; (void)n_in; (void)out_size; (void)ws_size;
    char* ws = (char*)d_ws;
    float* F0 = (float*)(ws + (0u  << 20));          // 4 MB each (32^3 x 32 f32)
    float* F1 = (float*)(ws + (4u  << 20));
    float* F2 = (float*)(ws + (8u  << 20));
    unsigned short* Hb = (unsigned short*)(ws + (12u << 20));  // act bf16 (2 MB)
    unsigned short* Wb = (unsigned short*)(ws + (14u << 20));  // packed wgt (1 MB)
    float* Zb = (float*)(ws + (15u << 20));                    // z 512x64
    float* Qb = (float*)(ws + (15u << 20) + (1u << 18));       // q 512x64

    const float* in0 = (const float*)d_in[0];
    auto P = [&](int i) { return (const float*)d_in[i]; };
    const float* NUL = nullptr;

    float* cur = F0; float* t1 = F1; float* t2 = F2;

    // ---------------- encoder ----------------
    conv_d0_kernel<<<(32768 * 32 + 255) / 256, 256, 0, stream>>>(in0, P(1), P(2), cur);
    resblock<32, 32>(d_in, Hb, Wb, cur, t1, t2, 3,  stream);
    resblock<32, 32>(d_in, Hb, Wb, cur, t1, t2, 12, stream);

    conv_launch<32, 16, 32, 64, 4, 2, MODE_RELU>(
        cur, P(21), P(22), NUL, NUL, NUL, NUL, NUL, NUL, t1, Hb, Wb, stream);
    { float* tmp = cur; cur = t1; t1 = tmp; }
    resblock<16, 64>(d_in, Hb, Wb, cur, t1, t2, 23, stream);
    resblock<16, 64>(d_in, Hb, Wb, cur, t1, t2, 32, stream);

    conv_launch<16, 8, 64, 128, 4, 2, MODE_RELU>(
        cur, P(41), P(42), NUL, NUL, NUL, NUL, NUL, NUL, t1, Hb, Wb, stream);
    { float* tmp = cur; cur = t1; t1 = tmp; }
    resblock<8, 128>(d_in, Hb, Wb, cur, t1, t2, 43, stream);
    resblock<8, 128>(d_in, Hb, Wb, cur, t1, t2, 52, stream);

    // enc_o: 8^3 x128 -> 8^3 x64, PReLU
    conv_launch<8, 8, 128, 64, 3, 1, MODE_PRELU>(
        cur, P(61), P(62), NUL, NUL, NUL, NUL, P(63), NUL, Zb, Hb, Wb, stream);

    // ---------------- vector quantization ----------------
    vq_kernel<<<8, 64, 0, stream>>>(Zb, P(64), Qb);

    // ---------------- decoder ----------------
    conv_launch<8, 8, 64, 128, 3, 1, MODE_PRELU>(
        Qb, P(65), P(66), NUL, NUL, NUL, NUL, P(67), NUL, cur, Hb, Wb, stream);
    resblock<8, 128>(d_in, Hb, Wb, cur, t1, t2, 68, stream);
    resblock<8, 128>(d_in, Hb, Wb, cur, t1, t2, 77, stream);

    convt_launch<8, 128, 64>(cur, P(86), P(87), t1, Hb, Wb, stream);
    { float* tmp = cur; cur = t1; t1 = tmp; }
    resblock<16, 64>(d_in, Hb, Wb, cur, t1, t2, 88, stream);
    resblock<16, 64>(d_in, Hb, Wb, cur, t1, t2, 97, stream);

    convt_launch<16, 64, 32>(cur, P(106), P(107), t1, Hb, Wb, stream);
    { float* tmp = cur; cur = t1; t1 = tmp; }
    resblock<32, 32>(d_in, Hb, Wb, cur, t1, t2, 108, stream);
    resblock<32, 32>(d_in, Hb, Wb, cur, t1, t2, 117, stream);

    convt_c1_kernel<<<(262144 + 255) / 256, 256, 0, stream>>>(
        cur, P(126), P(127), (float*)d_out);
}
